// Encoder_74517682586048
// MI455X (gfx1250) — compile-verified, matrix operand order checked by reading, not run on hardware
//
#include <hip/hip_runtime.h>
#include <hip/hip_bf16.h>

typedef __attribute__((ext_vector_type(16))) _Float16 v16h;
typedef __attribute__((ext_vector_type(8)))  float    v8f;

// ===========================================================================
// Weight prep: convert f32 weights -> f16, pre-swizzled into the exact WMMA
// B-fragment lane layout so the GEMM hot loop does pure b128 loads.
// Layout: out[mat][coltile][kstep][lane 0..31][16 f16]
//   lane: hs = lane>>4, mr = lane&15, krow = ks*32 + hs*8
//   elem i   = W[krow +    i][coltile*16 + mr]
//   elem 8+i = W[krow + 16+i][coltile*16 + mr]
// ===========================================================================
template <int K, int NCOL>
__global__ __launch_bounds__(32) void pack_weights(
    const float* __restrict__ m0, const float* __restrict__ m1,
    const float* __restrict__ m2, _Float16* __restrict__ out)
{
  const float* W = (blockIdx.y == 0) ? m0 : (blockIdx.y == 1) ? m1 : m2;
  const int ct   = blockIdx.x / (K / 32);
  const int ks   = blockIdx.x % (K / 32);
  const int lane = threadIdx.x;
  const int hs   = lane >> 4;
  const int mr   = lane & 15;
  const int krow = ks * 32 + hs * 8;
  const float* p = W + (size_t)krow * NCOL + ct * 16 + mr;
  _Float16* o = out +
      ((((size_t)blockIdx.y * (NCOL / 16) + ct) * (K / 32) + ks) * 32 + lane) * 16;
#pragma unroll
  for (int i = 0; i < 8; ++i) {
    o[i]     = (_Float16)p[(size_t)i * NCOL];
    o[8 + i] = (_Float16)p[(size_t)(16 + i) * NCOL];
  }
}

// ===========================================================================
// Node-level transform: 16-row node strip per block, one 16-col tile per wave,
// three GEMMs (rel0, rel1, self-loop) via v_wmma_f32_16x16x32_f16.
//  - A fragments: built once per block into LDS (wave ks builds k-step ks,
//    f32 load + optional relu + cvt to f16 in WMMA lane layout), then each
//    wave pulls them with ds_load_b128 pairs.
//  - B fragments: pre-swizzled f16 in global, 2x global_load_b128 each.
// ===========================================================================
template <int K, int NCOL, bool RELU>
__global__ __launch_bounds__((NCOL / 16) * 32) void node_xform(
    const float* __restrict__ x,         // [N, K]
    const _Float16* __restrict__ wpack,  // [3][NCOL/16][K/32][32][16]
    const float* __restrict__ bias,      // [NCOL]
    float* __restrict__ o0, float* __restrict__ o1, float* __restrict__ oagg)
{
  constexpr int KS = K / 32;
  __shared__ v16h ash[KS * 32];          // A fragments, WMMA lane layout

  const int lane = threadIdx.x & 31;
  const int wave = threadIdx.x >> 5;     // column tile
  const int row0 = blockIdx.x * 16;
  const int hs   = lane >> 4;
  const int mr   = lane & 15;

  // ---- build phase: wave ks converts k-step ks (block has >= KS waves) ----
  if (wave < KS) {
    const int    ks = wave;
    const float* ap = x + (size_t)(row0 + mr) * K + ks * 32 + hs * 8;
    v16h a;
#pragma unroll
    for (int i = 0; i < 8; ++i) {
      float lo = ap[i];
      float hi = ap[16 + i];
      if (RELU) { lo = fmaxf(lo, 0.0f); hi = fmaxf(hi, 0.0f); }
      a[i]     = (_Float16)lo;
      a[8 + i] = (_Float16)hi;
    }
    ash[ks * 32 + lane] = a;
  }
  __syncthreads();

  // ---- GEMM phase ---------------------------------------------------------
  v8f c0 = {}, c1 = {}, cl = {};
  constexpr size_t MATSTRIDE = (size_t)(NCOL / 16) * KS * 32 * 16;
  const _Float16* wb = wpack + ((size_t)wave * KS) * 32 * 16 + (size_t)lane * 16;

#pragma unroll
  for (int ks = 0; ks < KS; ++ks) {
    v16h a = ash[ks * 32 + lane];
    const _Float16* f = wb + (size_t)ks * 32 * 16;
    v16h b0 = *(const v16h*)(f);
    v16h b1 = *(const v16h*)(f + MATSTRIDE);
    v16h bl = *(const v16h*)(f + 2 * MATSTRIDE);
    c0 = __builtin_amdgcn_wmma_f32_16x16x32_f16(false, a, false, b0, (short)0, c0, false, false);
    c1 = __builtin_amdgcn_wmma_f32_16x16x32_f16(false, a, false, b1, (short)0, c1, false, false);
    cl = __builtin_amdgcn_wmma_f32_16x16x32_f16(false, a, false, bl, (short)0, cl, false, false);
  }

  const int   col = wave * 16 + mr;
  const float bv  = bias[col];
#pragma unroll
  for (int r = 0; r < 8; ++r) {
    size_t off = (size_t)(row0 + r + hs * 8) * NCOL + col;
    o0[off]   = c0[r];
    o1[off]   = c1[r];
    oagg[off] = cl[r] + bv;
  }
}

// ===========================================================================
// CSR build: counting sort of edges by dst (u32 atomics only).
// ===========================================================================
__global__ __launch_bounds__(256) void zero_u32(unsigned* p, int n) {
  int i = blockIdx.x * 256 + threadIdx.x;
  if (i < n) p[i] = 0u;
}

__global__ __launch_bounds__(256) void count_edges(
    const int* __restrict__ dst, unsigned* __restrict__ counts, int E) {
  int e = blockIdx.x * 256 + threadIdx.x;
  if (e < E) atomicAdd(&counts[dst[e]], 1u);
}

// Single-block shared-memory exclusive scan over n counts (n ~ 50k: 49 chunks
// of 1024, 10 Hillis-Steele steps each -- trivial runtime).
__global__ __launch_bounds__(1024) void scan_offsets(
    const unsigned* __restrict__ counts, unsigned* __restrict__ offsets,
    unsigned* __restrict__ cursor, int n)
{
  __shared__ unsigned s[1024];
  __shared__ unsigned carry;
  const int t = threadIdx.x;
  if (t == 0) carry = 0u;
  __syncthreads();
  for (int base = 0; base < n; base += 1024) {
    const int i = base + t;
    unsigned v = (i < n) ? counts[i] : 0u;
    s[t] = v;
    __syncthreads();
#pragma unroll
    for (int off = 1; off < 1024; off <<= 1) {
      unsigned u = (t >= off) ? s[t - off] : 0u;
      __syncthreads();
      s[t] += u;
      __syncthreads();
    }
    const unsigned excl  = s[t] - v + carry;
    const unsigned total = s[1023];
    if (i < n) { offsets[i] = excl; cursor[i] = excl; }
    __syncthreads();
    if (t == 0) carry += total;
    __syncthreads();
  }
  if (t == 0) offsets[n] = carry;
}

__global__ __launch_bounds__(256) void fill_edges(
    const int* __restrict__ src, const int* __restrict__ dst,
    const int* __restrict__ et, unsigned* __restrict__ cursor,
    unsigned* __restrict__ eidx, int E)
{
  int e = blockIdx.x * 256 + threadIdx.x;
  if (e >= E) return;
  unsigned pos = atomicAdd(&cursor[dst[e]], 1u);
  eidx[pos] = (unsigned)src[e] | ((unsigned)et[e] << 30);
}

// ===========================================================================
// Gather aggregation: one wave32 per destination node, 2-way unrolled with
// independent accumulators for load-level parallelism. Sums relation-selected
// source rows on top of the self-loop+bias init already in `io`. No float
// atomics anywhere; all gather traffic is L2-resident (x0/x1 = 25.6 MB).
// ===========================================================================
template <int F>  // 128 (VEC=4) or 64 (VEC=2)
__global__ __launch_bounds__(256) void aggregate(
    const float* __restrict__ x0, const float* __restrict__ x1,
    const unsigned* __restrict__ offsets, const unsigned* __restrict__ eidx,
    float* __restrict__ io, int N)
{
  constexpr int VEC = F / 32;
  int gid  = blockIdx.x * 256 + threadIdx.x;
  int node = gid >> 5;
  if (node >= N) return;
  int lane = gid & 31;

  const unsigned beg = offsets[node];
  const unsigned end = offsets[node + 1];

  float acc0[VEC], acc1[VEC];
#pragma unroll
  for (int i = 0; i < VEC; ++i) { acc0[i] = 0.0f; acc1[i] = 0.0f; }

  unsigned e = beg;
  for (; e + 2 <= end; e += 2) {
    const unsigned pk0 = eidx[e];
    const unsigned pk1 = eidx[e + 1];
    const float* r0 = ((pk0 >> 30) ? x1 : x0) + (size_t)(pk0 & 0x3FFFFFFFu) * F + lane * VEC;
    const float* r1 = ((pk1 >> 30) ? x1 : x0) + (size_t)(pk1 & 0x3FFFFFFFu) * F + lane * VEC;
    if (VEC == 4) {
      const float4 v0 = *(const float4*)r0;
      const float4 v1 = *(const float4*)r1;
      acc0[0] += v0.x; acc0[1] += v0.y; acc0[2] += v0.z; acc0[3] += v0.w;
      acc1[0] += v1.x; acc1[1] += v1.y; acc1[2] += v1.z; acc1[3] += v1.w;
    } else {
      const float2 v0 = *(const float2*)r0;
      const float2 v1 = *(const float2*)r1;
      acc0[0] += v0.x; acc0[1] += v0.y;
      acc1[0] += v1.x; acc1[1] += v1.y;
    }
  }
  if (e < end) {
    const unsigned pk = eidx[e];
    const float* r = ((pk >> 30) ? x1 : x0) + (size_t)(pk & 0x3FFFFFFFu) * F + lane * VEC;
    if (VEC == 4) {
      const float4 v = *(const float4*)r;
      acc0[0] += v.x; acc0[1] += v.y; acc0[2] += v.z; acc0[3] += v.w;
    } else {
      const float2 v = *(const float2*)r;
      acc0[0] += v.x; acc0[1] += v.y;
    }
  }

  float* o = io + (size_t)node * F + lane * VEC;
#pragma unroll
  for (int i = 0; i < VEC; ++i) o[i] += acc0[i] + acc1[i];
}

// ===========================================================================
extern "C" void kernel_launch(void* const* d_in, const int* in_sizes, int n_in,
                              void* d_out, int out_size, void* d_ws, size_t ws_size,
                              hipStream_t stream) {
  const float* feat  = (const float*)d_in[0];  // [N, 64]
  const float* W1    = (const float*)d_in[1];  // [2, 64, 128]
  const float* loop1 = (const float*)d_in[2];  // [64, 128]
  const float* b1    = (const float*)d_in[3];  // [128]
  const float* W2    = (const float*)d_in[4];  // [2, 128, 64]
  const float* loop2 = (const float*)d_in[5];  // [128, 64]
  const float* b2    = (const float*)d_in[6];  // [64]
  const int*   src   = (const int*)d_in[7];    // [E]
  const int*   dst   = (const int*)d_in[8];    // [E]
  const int*   et    = (const int*)d_in[9];    // [E]
  float*       out   = (float*)d_out;          // [N, 64]

  const int N = in_sizes[0] / 64;  // 50000 (multiple of 16)
  const int E = in_sizes[7];       // 800000

  // ---- workspace layout ----------------------------------------------------
  float* ws   = (float*)d_ws;
  float* x0   = ws;                      // [N,128] rel0 messages (layer1)
  float* x1   = x0 + (size_t)N * 128;    // [N,128] rel1 messages (layer1)
  float* hbuf = x1 + (size_t)N * 128;    // [N,128] selfinit -> aggregated h
  float* y0   = x0;                      // [N,64]  layer2 rel0 (aliases dead x0)
  float* y1   = x0 + (size_t)N * 64;     // [N,64]  layer2 rel1
  _Float16* wpack1 = (_Float16*)(hbuf + (size_t)N * 128);  // 3*8*2*32*16 = 24576
  _Float16* wpack2 = wpack1 + 24576;                       // 3*4*4*32*16 = 24576
  unsigned* counts  = (unsigned*)(wpack2 + 24576);
  unsigned* offsets = counts + N;        // N+1 entries (padded)
  unsigned* cursor  = offsets + N + 64;
  unsigned* eidx    = cursor + N;

  const int nblk = N / 16;                 // 3125 row strips
  const int eblk = (E + 255) / 256;        // per-edge kernels
  const int ablk = (N * 32 + 255) / 256;   // wave-per-node kernels
  const int zblk = (N + 255) / 256;

  // ---- weight prep (tiny) --------------------------------------------------
  pack_weights<64, 128><<<dim3(8 * 2, 3), 32, 0, stream>>>(
      W1, W1 + 64 * 128, loop1, wpack1);
  pack_weights<128, 64><<<dim3(4 * 4, 3), 32, 0, stream>>>(
      W2, W2 + 128 * 64, loop2, wpack2);

  // ---- CSR build (counting sort by dst, u32 atomics only) ------------------
  zero_u32<<<zblk, 256, 0, stream>>>(counts, N);
  count_edges<<<eblk, 256, 0, stream>>>(dst, counts, E);
  scan_offsets<<<1, 1024, 0, stream>>>(counts, offsets, cursor, N);
  fill_edges<<<eblk, 256, 0, stream>>>(src, dst, et, cursor, eidx, E);

  // ---- layer 1 -------------------------------------------------------------
  node_xform<64, 128, false><<<nblk, 256, 0, stream>>>(
      feat, wpack1, b1, x0, x1, hbuf);
  aggregate<128><<<ablk, 256, 0, stream>>>(x0, x1, offsets, eidx, hbuf, N);

  // ---- layer 2 (relu fused into A-fragment build) --------------------------
  node_xform<128, 64, true><<<nblk, 128, 0, stream>>>(
      hbuf, wpack2, b2, y0, y1, out);
  aggregate<64><<<ablk, 256, 0, stream>>>(y0, y1, offsets, eidx, out, N);
}
